// DGCNN_70153995813096
// MI455X (gfx1250) — matrix operand order repaired and checked by reading, not compiled, not run on hardware
//
#include <hip/hip_runtime.h>
#include <hip/hip_bf16.h>

// ---------------------------------------------------------------------------
// Types / WMMA helpers (gfx1250, wave32)
// ---------------------------------------------------------------------------
typedef __attribute__((ext_vector_type(8)))  float        v8f;
typedef __attribute__((ext_vector_type(16))) __bf16       v16bf;
typedef __attribute__((ext_vector_type(4)))  unsigned int ui4;
typedef __attribute__((ext_vector_type(8)))  int          i8v;
typedef __attribute__((ext_vector_type(4)))  int          i4v;

union BFrag { v16bf v; unsigned int u[8]; };

#if defined(__has_builtin)
#  if __has_builtin(__builtin_amdgcn_tensor_load_to_lds) && \
      __has_builtin(__builtin_amdgcn_s_wait_tensorcnt)
#    define HAVE_TDM 1
#  endif
#endif
#ifndef HAVE_TDM
#  define HAVE_TDM 0
#endif

__device__ inline unsigned short f2bf(float f) {
    unsigned int u = __float_as_uint(f);
    unsigned int r = (u + 0x7FFFu + ((u >> 16) & 1u)) >> 16;   // RNE
    return (unsigned short)r;
}
__device__ inline unsigned int pack2bf(float a, float b) {
    return (unsigned int)f2bf(a) | ((unsigned int)f2bf(b) << 16);
}

__device__ inline v8f wmma_bf16(const BFrag& a, const BFrag& b, v8f c) {
    // D(16x16,f32) = A(16x32,bf16) * B(32x16,bf16) + C
    return __builtin_amdgcn_wmma_f32_16x16x32_bf16(
        false, a.v, false, b.v, (short)0, c, false, false);
}

// A-matrix fragment (16x32 bf16) from LDS, rows packed as uint pairs.
// VGPR v holds K = 2v + hi*16 (+1), M = lane&15 (same K order as B).
__device__ inline BFrag load_fragA_lds(const unsigned int* rows, int rowStrideU, int ks) {
    const int lane = threadIdx.x & 31;
    const int m = lane & 15, hi = lane >> 4;
    BFrag f;
#pragma unroll
    for (int v = 0; v < 8; ++v) {
        int up = v + (hi << 3) + (ks << 4);
        f.u[v] = rows[m * rowStrideU + up];
    }
    return f;
}
// B-matrix fragment (32x16 bf16): N = lane&15, VGPR v holds K = 2v (+1), +16 hi lanes.
// rows[] laid out [16 rows][K/2 uints]; the 8 uints per (hi,ks) are contiguous -> b128s.
__device__ inline BFrag load_fragB_rows(const unsigned int* rows, int rowStrideU, int ks) {
    const int lane = threadIdx.x & 31;
    const int n = lane & 15, hi = lane >> 4;
    BFrag f;
#pragma unroll
    for (int v = 0; v < 8; ++v) {
        int up = v + (hi << 3) + (ks << 4);
        f.u[v] = rows[n * rowStrideU + up];
    }
    return f;
}

// ---------------------------------------------------------------------------
// TDM: 1-D contiguous global -> LDS copy via TENSOR_LOAD_TO_LDS (ISA ch.8).
// Descriptor: count=1, type=2, data_size=8B, tile_dim0 = nbytes/8, 1 row.
// ---------------------------------------------------------------------------
#if HAVE_TDM
__device__ inline void tdm_copy_to_lds(void* lds, const void* gsrc, unsigned int nbytes) {
    unsigned int cnt = nbytes >> 3;                       // elements of 8 bytes
    unsigned long long ga = (unsigned long long)(size_t)gsrc;
    unsigned int la = (unsigned int)(size_t)lds;          // LDS byte offset (addr[31:0])
    ui4 g0;
    g0[0] = 1u;                                           // count=1, user mode
    g0[1] = la;                                           // lds_addr
    g0[2] = (unsigned int)ga;                             // global_addr[31:0]
    g0[3] = (unsigned int)((ga >> 32) & 0x1FFFFFFull) | (2u << 30);  // addr[56:32], type=2
    i8v g1;
    g1[0] = (int)(3u << 16);                              // wg_mask=0, data_size=3 (8B)
    g1[1] = (int)((cnt & 0xFFFFu) << 16);                 // tensor_dim0[15:0]
    g1[2] = (int)(((cnt >> 16) & 0xFFFFu) | (1u << 16));  // tensor_dim0[31:16], tensor_dim1=1
    g1[3] = (int)((cnt & 0xFFFFu) << 16);                 // tile_dim0 = cnt
    g1[4] = 0;                                            // tile_dim1=0, tile_dim2=0
    g1[5] = (int)cnt;                                     // tensor_dim0_stride[31:0]
    g1[6] = 0;
    g1[7] = 0;
    i4v z = {0, 0, 0, 0};
#if __clang_major__ >= 23
    i8v z8 = {0, 0, 0, 0, 0, 0, 0, 0};
    __builtin_amdgcn_tensor_load_to_lds(g0, g1, z, z, z8, 0);
#else
    __builtin_amdgcn_tensor_load_to_lds(g0, g1, z, z, 0);
#endif
}
#endif

// ---------------------------------------------------------------------------
// Activation -> packed bf16 pairs: dst[p][ku] = (bf16(x[p][2ku]), bf16(x[p][2ku+1]))
// p runs over all 8*1024 rows; channels >= C zero-padded to 2*CPU.
// ---------------------------------------------------------------------------
__global__ void xconv(const float* __restrict__ src, int stride, int cofs,
                      int C, int CPU, unsigned int* __restrict__ dst) {
    int i = blockIdx.x * 256 + threadIdx.x;
    int total = 8 * 1024 * CPU;
    if (i >= total) return;
    int p = i / CPU, ku = i % CPU, c0 = 2 * ku;
    const float* r = src + (size_t)p * stride + cofs;
    float f0 = (c0     < C) ? r[c0]     : 0.f;
    float f1 = (c0 + 1 < C) ? r[c0 + 1] : 0.f;
    dst[i] = pack2bf(f0, f1);
}

// ---------------------------------------------------------------------------
// KNN distance GEMM: dist[b, r, c] = 2*x_r.x_c - |x_r|^2 - |x_c|^2
// A tile staged to LDS via TDM; A/B fragments are contiguous b128 loads.
// ---------------------------------------------------------------------------
template<int C, int CP>
__global__ __launch_bounds__(256)
void knn_gemm(const float* __restrict__ xsrc, int stride, int cofs,
              const unsigned int* __restrict__ xb16,
              float* __restrict__ dist) {
    constexpr int CPU = CP / 2;
    __shared__ float sq[1024];
    __shared__ unsigned int Au[16 * CPU];
    const int b = blockIdx.y, r0 = blockIdx.x * 16;
    const int tid = threadIdx.x;
    const float* xb = xsrc + (size_t)b * 1024 * stride + cofs;
    const unsigned int* xrows = xb16 + (size_t)b * 1024 * CPU;

#if HAVE_TDM
    if (tid == 0) tdm_copy_to_lds(Au, xrows + (size_t)r0 * CPU, 16 * CPU * 4);
#endif
    for (int p = tid; p < 1024; p += 256) {
        float s = 0.f;
        for (int c = 0; c < C; ++c) { float v = xb[(size_t)p * stride + c]; s += v * v; }
        sq[p] = s;
    }
#if HAVE_TDM
    __builtin_amdgcn_s_wait_tensorcnt(0);
#else
    for (int i = tid; i < 16 * CPU; i += 256) Au[i] = xrows[(size_t)r0 * CPU + i];
#endif
    __syncthreads();

    const int w = tid >> 5, lane = tid & 31;
    const int n = lane & 15, hi = lane >> 4;
    for (int ct = w; ct < 64; ct += 8) {
        const int c0 = ct * 16;
        v8f acc = {};
#pragma unroll
        for (int ks = 0; ks < CP / 32; ++ks) {
            BFrag a  = load_fragA_lds(Au, CPU, ks);
            BFrag bb = load_fragB_rows(xrows + (size_t)c0 * CPU, CPU, ks);
            acc = wmma_bf16(a, bb, acc);
        }
        const float sqc = sq[c0 + n];
        float* drow = dist + (size_t)(b * 1024 + r0) * 1024;
#pragma unroll
        for (int j = 0; j < 8; ++j) {
            int m = j + 8 * hi;
            drow[(size_t)m * 1024 + c0 + n] = 2.f * acc[j] - sq[r0 + m] - sqc;
        }
    }
}

// ---------------------------------------------------------------------------
// Top-K (K=20) per row, one wave32 per row; iterative argmax with shfl_xor.
// ---------------------------------------------------------------------------
__global__ __launch_bounds__(256)
void knn_topk(const float* __restrict__ dist, int* __restrict__ idx) {
    __shared__ float rowbuf[8][1024];
    const int b = blockIdx.y;
    const int w = threadIdx.x >> 5, lane = threadIdx.x & 31;
    const int row = blockIdx.x * 8 + w;
    const float4* dr4 = (const float4*)(dist + (size_t)(b * 1024 + row) * 1024);
    float4* rb4 = (float4*)rowbuf[w];
    for (int i = lane; i < 256; i += 32) rb4[i] = dr4[i];
    int* op = idx + (size_t)(b * 1024 + row) * 20;
    for (int t = 0; t < 20; ++t) {
        float best = -3.0e38f; int bi = 0x7fffffff;
        for (int i = lane; i < 1024; i += 32) {
            float v = rowbuf[w][i];
            if (v > best || (v == best && i < bi)) { best = v; bi = i; }
        }
#pragma unroll
        for (int off = 16; off >= 1; off >>= 1) {
            float ov = __shfl_xor(best, off, 32);
            int   oi = __shfl_xor(bi,   off, 32);
            if (ov > best || (ov == best && oi < bi)) { best = ov; bi = oi; }
        }
        if (lane == 0) op[t] = bi;
        rowbuf[w][bi] = -3.0e38f;   // all lanes store same value -> safe
    }
}

// ---------------------------------------------------------------------------
// EdgeConv: out[p,o] = max_k leaky(bn( e(p,k,:) . w[o,:] ))
// Point index = M-tile axis, k loops outside -> max-over-k is a register max over 20 k steps.
// ---------------------------------------------------------------------------
template<int C, int CK, int O, int NW>
__global__ __launch_bounds__(NW * 32)
void edge_conv(const float* __restrict__ xsrc, int stride, int cofs,
               const int* __restrict__ idx, const unsigned int* __restrict__ wbu,
               const float* __restrict__ scale, const float* __restrict__ bias,
               float* __restrict__ xout, int outOfs) {
    constexpr int CKU = CK / 2;
    constexpr int OTW = O / 16 / NW;
    __shared__ int nbi[20 * 16];
    __shared__ unsigned int Eu[16 * CKU];
    const int b = blockIdx.y, p0 = blockIdx.x * 16;
    const int tid = threadIdx.x, lane = tid & 31, w = tid >> 5;
    const int n = lane & 15, hi = lane >> 4;
    const float* xb = xsrc + (size_t)b * 1024 * stride + cofs;

    for (int i = tid; i < 320; i += NW * 32)
        nbi[i] = idx[(size_t)(b * 1024 + p0 + (i & 15)) * 20 + (i >> 4)];

    v8f mx[OTW];
    float sc[OTW], bs[OTW];
#pragma unroll
    for (int t = 0; t < OTW; ++t) {
        int o0 = (w + t * NW) * 16;
        sc[t] = scale[o0 + n]; bs[t] = bias[o0 + n];
#pragma unroll
        for (int j = 0; j < 8; ++j) mx[t][j] = -3.0e38f;
    }
    __syncthreads();

    for (int k = 0; k < 20; ++k) {
        for (int i = tid; i < 16 * CKU; i += NW * 32) {
            int m = i / CKU, ku = i % CKU, c0 = 2 * ku;
            int nb = nbi[k * 16 + m], ctr = p0 + m;
            int c1 = c0 + 1;
            float f0 = (c0 < C)     ? (xb[(size_t)nb * stride + c0] - xb[(size_t)ctr * stride + c0])
                     : (c0 < 2 * C) ?  xb[(size_t)ctr * stride + (c0 - C)] : 0.f;
            float f1 = (c1 < C)     ? (xb[(size_t)nb * stride + c1] - xb[(size_t)ctr * stride + c1])
                     : (c1 < 2 * C) ?  xb[(size_t)ctr * stride + (c1 - C)] : 0.f;
            Eu[i] = pack2bf(f0, f1);
        }
        __syncthreads();
#pragma unroll
        for (int t = 0; t < OTW; ++t) {
            int ot = w + t * NW;
            v8f acc = {};
#pragma unroll
            for (int ks = 0; ks < CK / 32; ++ks) {
                BFrag a  = load_fragA_lds(Eu, CKU, ks);
                BFrag bb = load_fragB_rows(wbu + (size_t)ot * 16 * CKU, CKU, ks);
                acc = wmma_bf16(a, bb, acc);
            }
#pragma unroll
            for (int j = 0; j < 8; ++j) {
                float h = acc[j] * sc[t] + bs[t];
                h = (h >= 0.f) ? h : 0.2f * h;
                mx[t][j] = fmaxf(mx[t][j], h);
            }
        }
        __syncthreads();
    }
#pragma unroll
    for (int t = 0; t < OTW; ++t) {
        int o0 = (w + t * NW) * 16;
#pragma unroll
        for (int j = 0; j < 8; ++j) {
            int m = j + 8 * hi;
            xout[(size_t)(b * 1024 + p0 + m) * 512 + outOfs + o0 + n] = mx[t][j];
        }
    }
}

// ---------------------------------------------------------------------------
// Per-point dense layer: h = leaky(bn(xcat(8192x512) @ w5^T(512x1024)))
// A tile (16 rows x 512 bf16, contiguous 16KB) staged via TDM.
// ---------------------------------------------------------------------------
__global__ __launch_bounds__(256)
void dense_gemm(const unsigned int* __restrict__ xcatb, const unsigned int* __restrict__ w5u,
                const float* __restrict__ scale, const float* __restrict__ bias,
                float* __restrict__ h) {
    __shared__ unsigned int Au[16 * 256];
    const int row0 = blockIdx.x * 16;
    const int tid = threadIdx.x, lane = tid & 31, w = tid >> 5;
    const int n = lane & 15, hi = lane >> 4;
#if HAVE_TDM
    if (tid == 0) tdm_copy_to_lds(Au, xcatb + (size_t)row0 * 256, 16 * 256 * 4);
    __builtin_amdgcn_s_wait_tensorcnt(0);
#else
    for (int i = tid; i < 16 * 256; i += 256) Au[i] = xcatb[(size_t)row0 * 256 + i];
#endif
    __syncthreads();
    for (int ot = w; ot < 64; ot += 8) {
        v8f acc = {};
#pragma unroll
        for (int ks = 0; ks < 16; ++ks) {
            BFrag a  = load_fragA_lds(Au, 256, ks);
            BFrag bb = load_fragB_rows(w5u + (size_t)ot * 16 * 256, 256, ks);
            acc = wmma_bf16(a, bb, acc);
        }
        int col = ot * 16 + n;
        float s = scale[col], bi = bias[col];
#pragma unroll
        for (int j = 0; j < 8; ++j) {
            int m = j + 8 * hi;
            float v = acc[j] * s + bi;
            v = (v >= 0.f) ? v : 0.2f * v;
            h[(size_t)(row0 + m) * 1024 + col] = v;
        }
    }
}

// ---------------------------------------------------------------------------
// Global max + mean pooling over N=1024 points.
// ---------------------------------------------------------------------------
__global__ __launch_bounds__(128)
void pool_maxmean(const float* __restrict__ h, float* __restrict__ g) {
    int b = blockIdx.y, ch = blockIdx.x * 128 + threadIdx.x;
    float mx = -3.0e38f, sm = 0.f;
    const float* hb = h + (size_t)b * 1024 * 1024 + ch;
    for (int n2 = 0; n2 < 1024; ++n2) {
        float v = hb[(size_t)n2 * 1024];
        mx = fmaxf(mx, v); sm += v;
    }
    g[b * 2048 + ch] = mx;
    g[b * 2048 + 1024 + ch] = sm * (1.f / 1024.f);
}

// ---------------------------------------------------------------------------
// Tiny FC layers (M=8 -> scalar path). mode 0: bn+leaky; mode 1: +bias only.
// ---------------------------------------------------------------------------
__global__ __launch_bounds__(128)
void fc_layer(const float* __restrict__ in, const float* __restrict__ wgt,
              const float* __restrict__ scale, const float* __restrict__ bias,
              const float* __restrict__ fcb, float* __restrict__ out,
              int In, int Out, int mode) {
    int b = blockIdx.y, o = blockIdx.x * 128 + threadIdx.x;
    if (o >= Out) return;
    const float* iv = in + (size_t)b * In;
    const float* wr = wgt + (size_t)o * In;
    float acc = 0.f;
    for (int k2 = 0; k2 < In; ++k2) acc += iv[k2] * wr[k2];
    float v;
    if (mode == 0) { v = acc * scale[o] + bias[o]; v = (v >= 0.f) ? v : 0.2f * v; }
    else           { v = acc + fcb[o]; }
    out[(size_t)b * Out + o] = v;
}

// ---------------------------------------------------------------------------
// BN fold: scale = g*rsqrt(v+eps), bias = b - m*scale
// ---------------------------------------------------------------------------
__global__ void bnfold(const float* g, const float* b, const float* m, const float* v,
                       float* scale, float* bias, int C) {
    int i = blockIdx.x * 256 + threadIdx.x;
    if (i < C) {
        float s = g[i] * rsqrtf(v[i] + 1e-5f);
        scale[i] = s; bias[i] = b[i] - m[i] * s;
    }
}

// f32 -> bf16 weight conversion, K padded to Kp (zeros).
__global__ void wconv(const float* src, unsigned short* dst, int O, int Kin, int Kp) {
    int i = blockIdx.x * 256 + threadIdx.x;
    if (i < O * Kp) {
        int k = i % Kp;
        dst[i] = (k < Kin) ? f2bf(src[(i / Kp) * Kin + k]) : (unsigned short)0;
    }
}

// ---------------------------------------------------------------------------
// Host launcher
// ---------------------------------------------------------------------------
extern "C" void kernel_launch(void* const* d_in, const int* in_sizes, int n_in,
                              void* d_out, int out_size, void* d_ws, size_t ws_size,
                              hipStream_t stream) {
    (void)in_sizes; (void)n_in; (void)out_size; (void)ws_size;
    const float* X    = (const float*)d_in[0];
    const float* W1   = (const float*)d_in[1];
    const float* BN1g = (const float*)d_in[2],  *BN1b = (const float*)d_in[3],
               * BN1m = (const float*)d_in[4],  *BN1v = (const float*)d_in[5];
    const float* W2   = (const float*)d_in[6];
    const float* BN2g = (const float*)d_in[7],  *BN2b = (const float*)d_in[8],
               * BN2m = (const float*)d_in[9],  *BN2v = (const float*)d_in[10];
    const float* W3   = (const float*)d_in[11];
    const float* BN3g = (const float*)d_in[12], *BN3b = (const float*)d_in[13],
               * BN3m = (const float*)d_in[14], *BN3v = (const float*)d_in[15];
    const float* W4   = (const float*)d_in[16];
    const float* BN4g = (const float*)d_in[17], *BN4b = (const float*)d_in[18],
               * BN4m = (const float*)d_in[19], *BN4v = (const float*)d_in[20];
    const float* W5   = (const float*)d_in[21];
    const float* BN5g = (const float*)d_in[22], *BN5b = (const float*)d_in[23],
               * BN5m = (const float*)d_in[24], *BN5v = (const float*)d_in[25];
    const float* FC1  = (const float*)d_in[26];
    const float* BN6g = (const float*)d_in[27], *BN6b = (const float*)d_in[28],
               * BN6m = (const float*)d_in[29], *BN6v = (const float*)d_in[30];
    const float* FC2  = (const float*)d_in[31];
    const float* BN7g = (const float*)d_in[32], *BN7b = (const float*)d_in[33],
               * BN7m = (const float*)d_in[34], *BN7v = (const float*)d_in[35];
    const float* FC3W = (const float*)d_in[36];
    const float* FC3B = (const float*)d_in[37];

    char* ws = (char*)d_ws;
    size_t off = 0;
    auto alloc = [&](size_t bytes) -> void* {
        void* p = ws + off;
        off += (bytes + 255) & ~(size_t)255;
        return p;
    };
    int*   IDX  = (int*)  alloc((size_t)8 * 1024 * 20 * 4);
    float* XCAT = (float*)alloc((size_t)8 * 1024 * 512 * 4);
    float* H    = (float*)alloc((size_t)8 * 1024 * 1024 * 4);   // also KNN dist scratch
    float* G    = (float*)alloc((size_t)8 * 2048 * 4);
    float* G1   = (float*)alloc((size_t)8 * 512 * 4);
    float* G2   = (float*)alloc((size_t)8 * 256 * 4);
    unsigned int* XB    = (unsigned int*)alloc((size_t)8192 * 64 * 4);   // packed bf16 acts
    unsigned int* XCATB = (unsigned int*)alloc((size_t)8192 * 256 * 4);  // packed bf16 xcat
    unsigned short* WB1 = (unsigned short*)alloc((size_t)64 * 32 * 2);
    unsigned short* WB2 = (unsigned short*)alloc((size_t)64 * 128 * 2);
    unsigned short* WB3 = (unsigned short*)alloc((size_t)128 * 128 * 2);
    unsigned short* WB4 = (unsigned short*)alloc((size_t)256 * 256 * 2);
    unsigned short* WB5 = (unsigned short*)alloc((size_t)1024 * 512 * 2);
    float* S1 = (float*)alloc(64 * 4);   float* B1 = (float*)alloc(64 * 4);
    float* S2 = (float*)alloc(64 * 4);   float* B2 = (float*)alloc(64 * 4);
    float* S3 = (float*)alloc(128 * 4);  float* B3 = (float*)alloc(128 * 4);
    float* S4 = (float*)alloc(256 * 4);  float* B4 = (float*)alloc(256 * 4);
    float* S5 = (float*)alloc(1024 * 4); float* B5 = (float*)alloc(1024 * 4);
    float* S6 = (float*)alloc(512 * 4);  float* B6 = (float*)alloc(512 * 4);
    float* S7 = (float*)alloc(256 * 4);  float* B7 = (float*)alloc(256 * 4);

    // BN folds + weight conversions
    bnfold<<<1, 256, 0, stream>>>(BN1g, BN1b, BN1m, BN1v, S1, B1, 64);
    bnfold<<<1, 256, 0, stream>>>(BN2g, BN2b, BN2m, BN2v, S2, B2, 64);
    bnfold<<<1, 256, 0, stream>>>(BN3g, BN3b, BN3m, BN3v, S3, B3, 128);
    bnfold<<<1, 256, 0, stream>>>(BN4g, BN4b, BN4m, BN4v, S4, B4, 256);
    bnfold<<<4, 256, 0, stream>>>(BN5g, BN5b, BN5m, BN5v, S5, B5, 1024);
    bnfold<<<2, 256, 0, stream>>>(BN6g, BN6b, BN6m, BN6v, S6, B6, 512);
    bnfold<<<1, 256, 0, stream>>>(BN7g, BN7b, BN7m, BN7v, S7, B7, 256);
    wconv<<<(64 * 32 + 255) / 256,    256, 0, stream>>>(W1, WB1, 64, 6, 32);
    wconv<<<(64 * 128 + 255) / 256,   256, 0, stream>>>(W2, WB2, 64, 128, 128);
    wconv<<<(128 * 128 + 255) / 256,  256, 0, stream>>>(W3, WB3, 128, 128, 128);
    wconv<<<(256 * 256 + 255) / 256,  256, 0, stream>>>(W4, WB4, 256, 256, 256);
    wconv<<<(1024 * 512 + 255) / 256, 256, 0, stream>>>(W5, WB5, 1024, 512, 512);

    dim3 gKnn(64, 8), gTop(128, 8), gEdge(64, 8);

    // Layer 1: x(.,3) -> xcat[:, 0:64)
    xconv<<<(8192 * 16 + 255) / 256, 256, 0, stream>>>(X, 3, 0, 3, 16, XB);
    knn_gemm<3, 32><<<gKnn, 256, 0, stream>>>(X, 3, 0, XB, H);
    knn_topk<<<gTop, 256, 0, stream>>>(H, IDX);
    edge_conv<3, 32, 64, 4><<<gEdge, 128, 0, stream>>>(
        X, 3, 0, IDX, (const unsigned int*)WB1, S1, B1, XCAT, 0);

    // Layer 2: xcat[:,0:64) -> xcat[:, 64:128)
    xconv<<<(8192 * 32 + 255) / 256, 256, 0, stream>>>(XCAT, 512, 0, 64, 32, XB);
    knn_gemm<64, 64><<<gKnn, 256, 0, stream>>>(XCAT, 512, 0, XB, H);
    knn_topk<<<gTop, 256, 0, stream>>>(H, IDX);
    edge_conv<64, 128, 64, 4><<<gEdge, 128, 0, stream>>>(
        XCAT, 512, 0, IDX, (const unsigned int*)WB2, S2, B2, XCAT, 64);

    // Layer 3: xcat[:,64:128) -> xcat[:, 128:256)
    xconv<<<(8192 * 32 + 255) / 256, 256, 0, stream>>>(XCAT, 512, 64, 64, 32, XB);
    knn_gemm<64, 64><<<gKnn, 256, 0, stream>>>(XCAT, 512, 64, XB, H);
    knn_topk<<<gTop, 256, 0, stream>>>(H, IDX);
    edge_conv<64, 128, 128, 8><<<gEdge, 256, 0, stream>>>(
        XCAT, 512, 64, IDX, (const unsigned int*)WB3, S3, B3, XCAT, 128);

    // Layer 4: xcat[:,128:256) -> xcat[:, 256:512)
    xconv<<<(8192 * 64 + 255) / 256, 256, 0, stream>>>(XCAT, 512, 128, 128, 64, XB);
    knn_gemm<128, 128><<<gKnn, 256, 0, stream>>>(XCAT, 512, 128, XB, H);
    knn_topk<<<gTop, 256, 0, stream>>>(H, IDX);
    edge_conv<128, 256, 256, 8><<<gEdge, 256, 0, stream>>>(
        XCAT, 512, 128, IDX, (const unsigned int*)WB4, S4, B4, XCAT, 256);

    // Dense 512->1024 + pooling + FCs
    xconv<<<(8192 * 256 + 255) / 256, 256, 0, stream>>>(XCAT, 512, 0, 512, 256, XCATB);
    dense_gemm<<<512, 256, 0, stream>>>(XCATB, (const unsigned int*)WB5, S5, B5, H);
    pool_maxmean<<<dim3(8, 8), 128, 0, stream>>>(H, G);
    fc_layer<<<dim3(4, 8), 128, 0, stream>>>(G,  FC1, S6, B6, nullptr, G1, 2048, 512, 0);
    fc_layer<<<dim3(2, 8), 128, 0, stream>>>(G1, FC2, S7, B7, nullptr, G2, 512, 256, 0);
    fc_layer<<<dim3(1, 8), 128, 0, stream>>>(G2, FC3W, nullptr, nullptr, FC3B,
                                             (float*)d_out, 256, 10, 1);
}
